// FFhGRU_9758165697057
// MI455X (gfx1250) — compile-verified, matrix operand order checked by reading, not run on hardware
//
#include <hip/hip_runtime.h>
#include <hip/hip_bf16.h>

// ---------------------------------------------------------------------------
// FFhGRU on MI455X (gfx1250): WMMA bf16 implicit-GEMM convs + async-LDS halos
// ---------------------------------------------------------------------------

#define Cch   32
#define Hh    96
#define Ww    96
#define Bb    4
#define HW    (Hh * Ww)              // 9216
#define NPIX  (Bb * HW)              // 36864
#define KK    15
#define TT    8
#define NGN   (Cch * HW)             // 294912 (GN count per sample)
#define NBN   NPIX                   // 36864  (BN count per channel)

typedef __bf16 v16bf __attribute__((ext_vector_type(16)));
typedef float  v8f   __attribute__((ext_vector_type(8)));

struct Frag { union { v16bf v; uint4 q[2]; }; };

__device__ __forceinline__ unsigned short f2bf(float f) {
    unsigned int u = __float_as_uint(f);
    unsigned int r = u + 0x7FFFu + ((u >> 16) & 1u);
    return (unsigned short)(r >> 16);
}
__device__ __forceinline__ float softplus_f(float x) {
    return (x > 20.f) ? x : log1pf(expf(x));
}
__device__ __forceinline__ float sigmoid_f(float x) {
    return 1.f / (1.f + expf(-x));
}
// Load one A/B fragment: row of 32 contiguous bf16 at rowbase; per-lane hi
// selects the K-halves per ISA 16-bit 16x32 layout (two aligned 16B chunks).
__device__ __forceinline__ void load_frag(Frag& f, const void* rowbase, int hi) {
    const char* p = (const char*)rowbase;
    f.q[0] = *(const uint4*)(p + hi * 16);
    f.q[1] = *(const uint4*)(p + 32 + hi * 16);
}
__device__ __forceinline__ v8f wmma_bf16(const Frag& a, const Frag& b, v8f c) {
    return __builtin_amdgcn_wmma_f32_16x16x32_bf16(false, a.v, false, b.v,
                                                   (short)0, c, false, false);
}
// 8 consecutive channels per lane (C-frag M = v + 8*hi) -> contiguous stores
__device__ __forceinline__ void store8(float* dst, const float* v) {
    float4 f0 = {v[0], v[1], v[2], v[3]};
    float4 f1 = {v[4], v[5], v[6], v[7]};
    *(float4*)dst = f0;
    *(float4*)(dst + 4) = f1;
}
__device__ __forceinline__ void store8bf(unsigned short* dst, const float* v) {
    union { unsigned short u[8]; uint4 q; } t;
#pragma unroll
    for (int i = 0; i < 8; ++i) t.u[i] = f2bf(v[i]);
    *(uint4*)dst = t.q;
}

// ---------------------------------------------------------------------------
// Weight repack: w[o][c][15][15] -> bf16 wpk[kk][o][c]; g_w -> bf16 (same order)
// ---------------------------------------------------------------------------
__global__ void repack_weights(const float* w_inh, const float* w_exc,
                               const float* g_w,
                               unsigned short* wpk_inh, unsigned short* wpk_exc,
                               unsigned short* wpk_g) {
    int i = blockIdx.x * 256 + threadIdx.x;
    if (i < 225 * 1024) {
        int kk = i >> 10, oc = i & 1023;
        int o_ = oc >> 5, c_ = oc & 31;
        int src = o_ * (Cch * KK * KK) + c_ * (KK * KK) + kk;
        wpk_inh[i] = f2bf(w_inh[src]);
        wpk_exc[i] = f2bf(w_exc[src]);
    }
    if (i < 8 * 1024) wpk_g[i] = f2bf(g_w[i]);
}

// ---------------------------------------------------------------------------
// Preproc: 7x7 conv 3->32 (pad 3) + softplus; NCHW in -> NHWC f32 + bf16 out
// ---------------------------------------------------------------------------
__global__ void preproc_kernel(const float* x, const float* pw, const float* pb,
                               float* xbn, unsigned short* xbn_bf) {
    int p = blockIdx.x * 256 + threadIdx.x;
    if (p >= NPIX) return;
    int xx = p % Ww, yy = (p / Ww) % Hh, b = p / HW;
    float acc[Cch];
#pragma unroll
    for (int o = 0; o < Cch; ++o) acc[o] = pb[o];
    for (int ci = 0; ci < 3; ++ci)
        for (int dy = 0; dy < 7; ++dy) {
            int y2 = yy + dy - 3;
            if (y2 < 0 || y2 >= Hh) continue;
            for (int dx = 0; dx < 7; ++dx) {
                int x2 = xx + dx - 3;
                if (x2 < 0 || x2 >= Ww) continue;
                float xv = x[((b * 3 + ci) * Hh + y2) * Ww + x2];
#pragma unroll
                for (int o = 0; o < Cch; ++o)
                    acc[o] = fmaf(xv, pw[((o * 3 + ci) * 7 + dy) * 7 + dx], acc[o]);
            }
        }
#pragma unroll
    for (int o = 0; o < Cch; ++o) {
        float v = softplus_f(acc[o]);
        xbn[(size_t)p * Cch + o]    = v;
        xbn_bf[(size_t)p * Cch + o] = f2bf(v);
    }
}

// ---------------------------------------------------------------------------
// 1x1 conv (optionally dual, summed) via WMMA.
// One wave -> ALL 32 out-ch x 16 px: B-fragment reused across both M-tiles.
// Optional GroupNorm sum/sumsq atomics per sample.
// ---------------------------------------------------------------------------
__global__ void conv1x1_dual_wmma(const unsigned short* wpk, const float* gb,
                                  int idx0, int idx1,
                                  const unsigned short* x0,
                                  const unsigned short* x1,
                                  float* out, unsigned short* out_bf,
                                  float* gnstats, int gate) {
    int tid = threadIdx.x, lane = tid & 31;
    int ptile = blockIdx.x * 8 + (tid >> 5);  // 0..2303
    int n = lane & 15, hi = lane >> 4;
    size_t pix = (size_t)ptile * 16 + n;      // B-matrix column (pixel)
    int b = (ptile * 16) / HW;

    v8f acc0 = {0.f, 0.f, 0.f, 0.f, 0.f, 0.f, 0.f, 0.f};   // out ch 0..15
    v8f acc1 = {0.f, 0.f, 0.f, 0.f, 0.f, 0.f, 0.f, 0.f};   // out ch 16..31
    {
        Frag a0, a1, bf;
        const char* wb = (const char*)wpk + (size_t)idx0 * 2048;
        load_frag(a0, wb + (size_t)n * 64, hi);
        load_frag(a1, wb + (size_t)(16 + n) * 64, hi);
        load_frag(bf, (const char*)x0 + pix * 64, hi);
        acc0 = wmma_bf16(a0, bf, acc0);
        acc1 = wmma_bf16(a1, bf, acc1);
    }
    if (idx1 >= 0) {
        Frag a0, a1, bf;
        const char* wb = (const char*)wpk + (size_t)idx1 * 2048;
        load_frag(a0, wb + (size_t)n * 64, hi);
        load_frag(a1, wb + (size_t)(16 + n) * 64, hi);
        load_frag(bf, (const char*)x1 + pix * 64, hi);
        acc0 = wmma_bf16(a0, bf, acc0);
        acc1 = wmma_bf16(a1, bf, acc1);
    }
    float s = 0.f, s2 = 0.f;
#pragma unroll
    for (int m = 0; m < 2; ++m) {
        int chbase = m * 16 + 8 * hi;
        float vals[8];
#pragma unroll
        for (int v = 0; v < 8; ++v) {
            int ch = chbase + v;
            float bias = gb[idx0 * Cch + ch] +
                         ((idx1 >= 0) ? gb[idx1 * Cch + ch] : 0.f);
            float val = ((m == 0) ? acc0[v] : acc1[v]) + bias;
            vals[v] = val;
            s += val; s2 += val * val;
        }
        store8(&out[pix * Cch + chbase], vals);
        if (out_bf) store8bf(&out_bf[pix * Cch + chbase], vals);
    }
    if (gnstats) {
#pragma unroll
        for (int mk = 1; mk < 32; mk <<= 1) {
            s  += __shfl_xor(s,  mk, 32);
            s2 += __shfl_xor(s2, mk, 32);
        }
        if (lane == 0) {
            atomicAdd(&gnstats[(gate * 4 + b) * 2 + 0], s);
            atomicAdd(&gnstats[(gate * 4 + b) * 2 + 1], s2);
        }
    }
}

// ---------------------------------------------------------------------------
// 15x15 conv 32->32 implicit GEMM. Block = 4 waves (4 output rows x 32 cols).
// Each wave computes a 32ch x 32px tile: 4 WMMAs per tap, A/B frags reused 2x.
// Halo (18 x 46 x 32ch bf16 = 52 KB) staged via CDNA5 async global->LDS loads.
// Per-channel BatchNorm sum/sumsq atomics.
// ---------------------------------------------------------------------------
#define TCOLS 46
#define TROWS 18
__device__ __forceinline__ void conv15_epi(v8f accP0, v8f accP1, int chbase,
                                           const float* bias, float* out,
                                           size_t p0, int n,
                                           float* bnsum, float* bnsq) {
    float vp0[8], vp1[8];
#pragma unroll
    for (int v = 0; v < 8; ++v) {
        float bb = bias[chbase + v];
        vp0[v] = accP0[v] + bb;
        vp1[v] = accP1[v] + bb;
    }
    store8(&out[(p0 + n) * Cch + chbase], vp0);
    store8(&out[(p0 + 16 + n) * Cch + chbase], vp1);
#pragma unroll
    for (int v = 0; v < 8; ++v) {
        float s = vp0[v] + vp1[v];
        float s2 = vp0[v] * vp0[v] + vp1[v] * vp1[v];
#pragma unroll
        for (int mk = 1; mk < 16; mk <<= 1) {   // reduce across the 16 n-lanes
            s  += __shfl_xor(s,  mk, 32);
            s2 += __shfl_xor(s2, mk, 32);
        }
        if (n == 0) {
            atomicAdd(&bnsum[chbase + v], s);
            atomicAdd(&bnsq[chbase + v], s2);
        }
    }
}

__global__ void conv15_wmma(const unsigned short* src, const unsigned short* wpk,
                            const float* bias, float* out,
                            float* bnsum, float* bnsq) {
    __shared__ __align__(16) unsigned char tile[TROWS * TCOLS * 64]; // 52992 B
    int tid = threadIdx.x;                 // 128 threads
    int blk = blockIdx.x;                  // 4 * 24 * 3 = 288
    int xbi = blk % 3, t2 = blk / 3;
    int yblk = t2 % (Hh / 4), b = t2 / (Hh / 4);
    int xb = xbi * 32, ybase = yblk * 4;

    // ---- stage halo into LDS via per-lane async b128 copies ----
    const int NPXT = TROWS * TCOLS;        // 828 pixels x 64B
    for (int i = tid; i < NPXT; i += 128) {
        int tr = i / TCOLS, tc = i % TCOLS;
        int yy = ybase + tr - 7, xx = xb + tc - 7;
        unsigned int lds = (unsigned int)(size_t)(&tile[i * 64]);
        if (yy >= 0 && yy < Hh && xx >= 0 && xx < Ww) {
            const unsigned short* g = src + ((size_t)((b * Hh + yy) * Ww + xx)) * Cch;
            asm volatile("global_load_async_to_lds_b128 %0, %1, off offset:0"
                         :: "v"(lds), "v"(g) : "memory");
            asm volatile("global_load_async_to_lds_b128 %0, %1, off offset:16"
                         :: "v"(lds), "v"(g) : "memory");
            asm volatile("global_load_async_to_lds_b128 %0, %1, off offset:32"
                         :: "v"(lds), "v"(g) : "memory");
            asm volatile("global_load_async_to_lds_b128 %0, %1, off offset:48"
                         :: "v"(lds), "v"(g) : "memory");
        } else {
            uint4 z = {0u, 0u, 0u, 0u};
            *(uint4*)&tile[i * 64 +  0] = z;
            *(uint4*)&tile[i * 64 + 16] = z;
            *(uint4*)&tile[i * 64 + 32] = z;
            *(uint4*)&tile[i * 64 + 48] = z;
        }
    }
    asm volatile("s_wait_asynccnt 0" ::: "memory");
    __syncthreads();

    // ---- implicit GEMM: 225 taps x 4 chained K=32 WMMAs ----
    int wv = tid >> 5, lane = tid & 31;
    int y = ybase + wv;
    int n = lane & 15, hi = lane >> 4;

    v8f acc00 = {0.f, 0.f, 0.f, 0.f, 0.f, 0.f, 0.f, 0.f};  // ch 0..15,  px 0..15
    v8f acc01 = acc00;                                      // ch 0..15,  px 16..31
    v8f acc10 = acc00;                                      // ch 16..31, px 0..15
    v8f acc11 = acc00;                                      // ch 16..31, px 16..31
    for (int ky = 0; ky < KK; ++ky) {
        int tr = wv + ky;
        for (int kx = 0; kx < KK; ++kx) {
            int kk = ky * KK + kx;
            Frag a0, a1, b0, b1;
            const char* wb = (const char*)wpk + (size_t)kk * 2048;
            load_frag(a0, wb + (size_t)n * 64, hi);
            load_frag(a1, wb + (size_t)(16 + n) * 64, hi);
            load_frag(b0, &tile[(size_t)(tr * TCOLS + n + kx) * 64], hi);
            load_frag(b1, &tile[(size_t)(tr * TCOLS + 16 + n + kx) * 64], hi);
            acc00 = wmma_bf16(a0, b0, acc00);
            acc10 = wmma_bf16(a1, b0, acc10);
            acc01 = wmma_bf16(a0, b1, acc01);
            acc11 = wmma_bf16(a1, b1, acc11);
        }
    }

    size_t p0 = (size_t)(b * Hh + y) * Ww + xb;   // pixel base
    conv15_epi(acc00, acc01,      8 * hi, bias, out, p0, n, bnsum, bnsq);
    conv15_epi(acc10, acc11, 16 + 8 * hi, bias, out, p0, n, bnsum, bnsq);
}

// ---------------------------------------------------------------------------
// Stats: zero + finalize (GN per-sample mean/inv; BN folded scale/shift)
// stats layout (floats): [0,24) gn sum/sq; [32,160) bn sum/sq (2x(32+32));
// [160,184) gn mean/inv; [192,320) bn scale/shift (2x(32+32))
// ---------------------------------------------------------------------------
__global__ void zero_stats_kernel(float* st) {
    int t = threadIdx.x;
    st[t] = 0.f; st[t + 256] = 0.f;
}
__global__ void finalize_stats(float* st, const float* bn_g, const float* bn_b) {
    int t = threadIdx.x;   // 64
    if (t < 12) {
        float s = st[t * 2], s2 = st[t * 2 + 1];
        float mean = s / (float)NGN;
        float var  = s2 / (float)NGN - mean * mean;
        st[160 + t * 2]     = mean;
        st[160 + t * 2 + 1] = rsqrtf(var + 1e-5f);
    }
    {
        int which = t >> 5, c = t & 31;
        float s  = st[32 + which * 64 + c];
        float s2 = st[32 + which * 64 + 32 + c];
        float mean = s / (float)NBN;
        float var  = s2 / (float)NBN - mean * mean;
        float sc = bn_g[which * Cch + c] * rsqrtf(var + 1e-3f);
        st[192 + which * 64 + c]      = sc;
        st[192 + which * 64 + 32 + c] = bn_b[which * Cch + c] - mean * sc;
    }
}

// ---------------------------------------------------------------------------
// Pointwise kernels
// ---------------------------------------------------------------------------
__global__ void pw_gated_exc(const float* tg0, const float* exc, const float* st,
                             const float* gbn_g, const float* gbn_b,
                             unsigned short* gex_bf) {
    size_t i = (size_t)blockIdx.x * 256 + threadIdx.x;
    int c = (int)(i & 31);
    int b = (int)((i >> 5) / HW);
    float z = (tg0[i] - st[160 + b * 2]) * st[160 + b * 2 + 1];
    float g = sigmoid_f(z * gbn_g[c] + gbn_b[c]);
    gex_bf[i] = f2bf(g * exc[i]);
}

__global__ void pw_update_inh(const float* tconv, const float* xbn,
                              float* inh, unsigned short* inh_bf,
                              const float* tg1, const float* st,
                              const float* alpha, const float* mu,
                              const float* gbn_g, const float* gbn_b) {
    size_t i = (size_t)blockIdx.x * 256 + threadIdx.x;
    int c = (int)(i & 31);
    int b = (int)((i >> 5) / HW);
    float intx = st[192 + c] * tconv[i] + st[224 + c];          // BN0
    float ih   = inh[i];                                        // gated_inh (old)
    float hat  = softplus_f(xbn[i] - intx * (alpha[c] * ih + mu[c]));
    float z = (tg1[i] - st[160 + 8 + b * 2]) * st[160 + 8 + b * 2 + 1];
    float g = sigmoid_f(z * gbn_g[Cch + c] + gbn_b[Cch + c]);
    float nw = (1.f - g) * ih + g * hat;
    inh[i] = nw; inh_bf[i] = f2bf(nw);
}

__global__ void pw_update_exc(const float* tconv, const float* inh,
                              float* exc, unsigned short* exc_bf,
                              const float* tg2, const float* st,
                              const float* kappa, const float* gamma_,
                              const float* gbn_g, const float* gbn_b) {
    size_t i = (size_t)blockIdx.x * 256 + threadIdx.x;
    int c = (int)(i & 31);
    int b = (int)((i >> 5) / HW);
    float intx = st[256 + c] * tconv[i] + st[288 + c];          // BN1
    float ih   = inh[i];                                        // updated inhibition
    float hat  = softplus_f(intx * (kappa[c] * ih + gamma_[c]));
    float z = (tg2[i] - st[160 + 16 + b * 2]) * st[160 + 16 + b * 2 + 1];
    float g = sigmoid_f(z * gbn_g[2 * Cch + c] + gbn_b[2 * Cch + c]);
    float e = exc[i];
    float nw = (1.f - g) * e + g * hat;
    exc[i] = nw; exc_bf[i] = f2bf(nw);
}

// NHWC -> NCHW final output
__global__ void transpose_out(const float* exc, float* out) {
    size_t i = (size_t)blockIdx.x * 256 + threadIdx.x;
    int x = (int)(i % Ww);
    int y = (int)((i / Ww) % Hh);
    int c = (int)((i / HW) % Cch);
    int b = (int)(i / (HW * Cch));
    out[i] = exc[((size_t)((b * Hh + y) * Ww + x)) * Cch + c];
}

// ---------------------------------------------------------------------------
extern "C" void kernel_launch(void* const* d_in, const int* in_sizes, int n_in,
                              void* d_out, int out_size, void* d_ws, size_t ws_size,
                              hipStream_t stream) {
    const float* x       = (const float*)d_in[0];
    const float* pre_w   = (const float*)d_in[1];
    const float* pre_b   = (const float*)d_in[2];
    const float* g_w     = (const float*)d_in[3];
    const float* g_b     = (const float*)d_in[4];
    const float* w_exc   = (const float*)d_in[5];
    const float* w_exc_b = (const float*)d_in[6];
    const float* w_inh   = (const float*)d_in[7];
    const float* w_inh_b = (const float*)d_in[8];
    const float* alpha   = (const float*)d_in[9];
    const float* mu      = (const float*)d_in[10];
    const float* gamma_  = (const float*)d_in[11];
    const float* kappa   = (const float*)d_in[12];
    const float* gbn_g   = (const float*)d_in[13];
    const float* gbn_b   = (const float*)d_in[14];
    const float* bn_g    = (const float*)d_in[15];
    const float* bn_b    = (const float*)d_in[16];

    char* wsb = (char*)d_ws;
    size_t off = 0;
    auto take = [&](size_t bytes) -> char* {
        char* p = wsb + off;
        off += (bytes + 255) & ~(size_t)255;
        return p;
    };
    const size_t F32 = (size_t)NPIX * Cch * 4;
    const size_t BF16 = (size_t)NPIX * Cch * 2;
    float*          xbn     = (float*)take(F32);
    float*          inh     = (float*)take(F32);
    float*          exc     = (float*)take(F32);
    float*          tg0     = (float*)take(F32);
    float*          tg1     = (float*)take(F32);
    float*          tg2     = (float*)take(F32);
    float*          tconv   = (float*)take(F32);
    unsigned short* xbn_bf  = (unsigned short*)take(BF16);
    unsigned short* inh_bf  = (unsigned short*)take(BF16);
    unsigned short* exc_bf  = (unsigned short*)take(BF16);
    unsigned short* gex_bf  = (unsigned short*)take(BF16);
    unsigned short* wpk_inh = (unsigned short*)take((size_t)225 * 1024 * 2);
    unsigned short* wpk_exc = (unsigned short*)take((size_t)225 * 1024 * 2);
    unsigned short* wpk_g   = (unsigned short*)take((size_t)8 * 1024 * 2);
    float*          st      = (float*)take(512 * 4);

    const dim3 B256(256), B128(128), B64(64);
    const int G1x1 = 288;                 // 2304 pixel-tiles / 8 waves per block
    const int G15  = Bb * (Hh / 4) * 3;   // 288
    const int GPW  = (NPIX * Cch) / 256;  // 4608
    (void)in_sizes; (void)n_in; (void)out_size; (void)ws_size;

    repack_weights<<<900, B256, 0, stream>>>(w_inh, w_exc, g_w, wpk_inh, wpk_exc, wpk_g);
    preproc_kernel<<<NPIX / 256, B256, 0, stream>>>(x, pre_w, pre_b, xbn, xbn_bf);
    // init states: inhibition = g6*xbn + b6 ; excitation = g7*xbn + b7
    conv1x1_dual_wmma<<<G1x1, B256, 0, stream>>>(wpk_g, g_b, 6, -1, xbn_bf, nullptr,
                                                 inh, inh_bf, nullptr, 0);
    conv1x1_dual_wmma<<<G1x1, B256, 0, stream>>>(wpk_g, g_b, 7, -1, xbn_bf, nullptr,
                                                 exc, exc_bf, nullptr, 0);

    for (int t = 0; t < TT; ++t) {
        zero_stats_kernel<<<1, B256, 0, stream>>>(st);
        // att gate pre-act = g0*xbn + g1*exc (+biases), GN0 stats
        conv1x1_dual_wmma<<<G1x1, B256, 0, stream>>>(wpk_g, g_b, 0, 1, xbn_bf, exc_bf,
                                                     tg0, nullptr, st, 0);
        finalize_stats<<<1, B64, 0, stream>>>(st, bn_g, bn_b);
        pw_gated_exc<<<GPW, B256, 0, stream>>>(tg0, exc, st, gbn_g, gbn_b, gex_bf);
        // inh interaction: 15x15 conv of gated_exc, BN0 stats
        conv15_wmma<<<G15, B128, 0, stream>>>(gex_bf, wpk_inh, w_inh_b, tconv,
                                              st + 32, st + 64);
        // inh gate pre-act = g2*xbn + g3*inh(old), GN1 stats
        conv1x1_dual_wmma<<<G1x1, B256, 0, stream>>>(wpk_g, g_b, 2, 3, xbn_bf, inh_bf,
                                                     tg1, nullptr, st, 1);
        // exc gate pre-act = g4*inh(old) + g5*gated_exc, GN2 stats
        conv1x1_dual_wmma<<<G1x1, B256, 0, stream>>>(wpk_g, g_b, 4, 5, inh_bf, gex_bf,
                                                     tg2, nullptr, st, 2);
        finalize_stats<<<1, B64, 0, stream>>>(st, bn_g, bn_b);
        pw_update_inh<<<GPW, B256, 0, stream>>>(tconv, xbn, inh, inh_bf, tg1, st,
                                                alpha, mu, gbn_g, gbn_b);
        // exc interaction: 15x15 conv of new inhibition, BN1 stats
        conv15_wmma<<<G15, B128, 0, stream>>>(inh_bf, wpk_exc, w_exc_b, tconv,
                                              st + 96, st + 128);
        finalize_stats<<<1, B64, 0, stream>>>(st, bn_g, bn_b);
        pw_update_exc<<<GPW, B256, 0, stream>>>(tconv, inh, exc, exc_bf, tg2, st,
                                                kappa, gamma_, gbn_g, gbn_b);
    }
    transpose_out<<<GPW, B256, 0, stream>>>(exc, (float*)d_out);
}
